// VectorQuantizer2_19963007992606
// MI455X (gfx1250) — matrix operand order repaired and checked by reading, not compiled, not run on hardware
//
#include <hip/hip_runtime.h>

// ---------- types for WMMA fragments ----------
typedef __attribute__((ext_vector_type(16))) __bf16 bf16x16;
typedef __attribute__((ext_vector_type(8)))  float  f32x8;
typedef __attribute__((ext_vector_type(4)))  float  f32x4;

union Frag {
  bf16x16 v;
  f32x4   q[2];   // two 16B halves for b128 loads
};

// round-to-nearest-even f32 -> bf16 (bit manipulation; avoids scalar __bf16 ops)
__device__ __forceinline__ unsigned short f2bf(float x) {
  unsigned u = __float_as_uint(x);
  u += 0x7FFFu + ((u >> 16) & 1u);
  return (unsigned short)(u >> 16);
}
__device__ __forceinline__ float bf2f(unsigned short h) {
  return __uint_as_float(((unsigned)h) << 16);
}

// ---------------------------------------------------------------------------
// Kernel 0: codebook [1024,256] f32 -> bf16 hi/lo split arrays + ||e||^2 (f32)
// ---------------------------------------------------------------------------
__global__ void vq_prep(const float* __restrict__ cb,
                        unsigned short* __restrict__ chi,
                        unsigned short* __restrict__ clo,
                        float* __restrict__ enorm) {
  const int code = blockIdx.x;     // 1024 blocks
  const int d    = threadIdx.x;    // 256 threads
  float v = cb[code * 256 + d];
  unsigned short h = f2bf(v);
  chi[code * 256 + d] = h;
  clo[code * 256 + d] = f2bf(v - bf2f(h));

  float s = v * v;
  #pragma unroll
  for (int m = 16; m >= 1; m >>= 1) s += __shfl_xor(s, m, 32);
  __shared__ float red[8];
  if ((threadIdx.x & 31) == 0) red[threadIdx.x >> 5] = s;
  __syncthreads();
  if (threadIdx.x == 0) {
    float t = 0.f;
    #pragma unroll
    for (int i = 0; i < 8; ++i) t += red[i];
    enorm[code] = t;
  }
}

// ---------------------------------------------------------------------------
// Kernel 1: main. 1024 blocks x 128 threads (4 wave32). Each block owns 64
// consecutive rows n = blockIdx.x*64 .. +63 (n = bt*1024 + hw), each wave a
// 16-row stripe. Distance GEMM via split-bf16 (hi/lo) WMMA, running argmin,
// then gather + z_q store + loss partial.
// ---------------------------------------------------------------------------
#define AS 264   // LDS row stride (ushort) for A arrays: 528B rows, 16B aligned

__global__ void __launch_bounds__(128)
vq_main(const float* __restrict__ z, const float* __restrict__ cb,
        const unsigned short* __restrict__ chi,
        const unsigned short* __restrict__ clo,
        const float* __restrict__ enorm,
        float* __restrict__ zq, int* __restrict__ idx_out,
        float* __restrict__ partials) {
  __shared__ unsigned short Ahi[64 * AS];   // 33,792 B
  __shared__ unsigned short Alo[64 * AS];   // 33,792 B
  __shared__ float eLDS[1024];              //  4,096 B
  __shared__ int   idxLDS[64];
  __shared__ float redLDS[4];

  const int t   = threadIdx.x;
  const int n0  = blockIdx.x * 64;
  const int bt  = n0 >> 10;        // which (b,t) slab
  const int hw0 = n0 & 1023;       // hw offset inside slab

  // ---- stage z tile (64 rows x 256 dims) as bf16 hi/lo in LDS ----
  // z memory layout [bt, c, hw]: row n dim d lives at (bt*256+d)*1024 + hw.
  for (int i = 0; i < 128; ++i) {
    int li = i * 128 + t;
    int d  = li >> 6;              // 0..255
    int hw = li & 63;              // 0..63
    float v = z[(bt * 256 + d) * 1024 + hw0 + hw];   // 256B-coalesced
    unsigned short h = f2bf(v);
    Ahi[hw * AS + d] = h;
    Alo[hw * AS + d] = f2bf(v - bf2f(h));
  }
  for (int i = 0; i < 8; ++i) eLDS[i * 128 + t] = enorm[i * 128 + t];
  __syncthreads();

  const int w    = t >> 5;         // wave id 0..3
  const int lane = t & 31;
  const int nl   = lane & 15;      // N within 16-wide code tile (C layout)
  const int half = lane >> 4;

  // A-fragment base: lane M = nl, row = w*16+nl; ISA A layout -> two 16B
  // segments per 32-K chunk at K-offsets {half*8, 16+half*8}.
  const unsigned short* ahp = &Ahi[(w * 16 + nl) * AS + half * 8];
  const unsigned short* alp = &Alo[(w * 16 + nl) * AS + half * 8];

  float best[8];
  int   bidx[8];
  #pragma unroll
  for (int r = 0; r < 8; ++r) { best[r] = 3.4e38f; bidx[r] = 0; }

  for (int tile = 0; tile < 64; ++tile) {
    const int code = tile * 16 + nl;
    // B-fragment: lane N = nl, two 8-K segments per lane at K-offsets {half*8, 16+half*8}.
    const unsigned short* bhp = &chi[code * 256 + half * 8];
    const unsigned short* blp = &clo[code * 256 + half * 8];

    f32x8 acc = {};
    #pragma unroll
    for (int kc = 0; kc < 8; ++kc) {     // K = 256 in chunks of 32
      Frag ah, al, bh, bl;
      ah.q[0] = *(const f32x4*)(ahp + kc * 32);
      ah.q[1] = *(const f32x4*)(ahp + kc * 32 + 16);
      al.q[0] = *(const f32x4*)(alp + kc * 32);
      al.q[1] = *(const f32x4*)(alp + kc * 32 + 16);
      bh.q[0] = *(const f32x4*)(bhp + kc * 32);
      bh.q[1] = *(const f32x4*)(bhp + kc * 32 + 16);
      bl.q[0] = *(const f32x4*)(blp + kc * 32);
      bl.q[1] = *(const f32x4*)(blp + kc * 32 + 16);
      // split-bf16 fp32 emulation: zh*eh + zh*el + zl*eh, f32 accumulate
      acc = __builtin_amdgcn_wmma_f32_16x16x32_bf16(false, ah.v, false, bh.v,
                                                    (short)0, acc, false, false);
      acc = __builtin_amdgcn_wmma_f32_16x16x32_bf16(false, ah.v, false, bl.v,
                                                    (short)0, acc, false, false);
      acc = __builtin_amdgcn_wmma_f32_16x16x32_bf16(false, al.v, false, bh.v,
                                                    (short)0, acc, false, false);
    }
    // score = ||e||^2 - 2 z.e  (||z||^2 is row-constant, irrelevant to argmin)
    const float en = eLDS[code];
    #pragma unroll
    for (int r = 0; r < 8; ++r) {
      float s = en - 2.0f * acc[r];
      if (s < best[r]) { best[r] = s; bidx[r] = code; }  // tiles ascend -> first-min
    }
  }

  // cross-lane argmin within each 16-lane half (C layout: lanes hold N,
  // VGPR slot r holds M=r (lanes<16) / M=r+8 (lanes>=16)); tie -> lower idx.
  #pragma unroll
  for (int r = 0; r < 8; ++r) {
    float b  = best[r];
    int   bi = bidx[r];
    #pragma unroll
    for (int m = 1; m < 16; m <<= 1) {
      float ob = __shfl_xor(b, m, 32);
      int   oi = __shfl_xor(bi, m, 32);
      if (ob < b || (ob == b && oi < bi)) { b = ob; bi = oi; }
    }
    if (nl == 0) {
      int row = w * 16 + half * 8 + r;
      idxLDS[row] = bi;
      idx_out[n0 + row] = bi;
    }
  }
  __syncthreads();

  // ---- pass 2: z_q gather/store ([b,t,c,h,w] layout) + loss partial ----
  float lsum = 0.f;
  const int hw  = t & 63;
  const int cb0 = t >> 6;
  const int code = idxLDS[hw];
  for (int i = 0; i < 128; ++i) {
    int c = i * 2 + cb0;
    float q  = cb[code * 256 + c];                    // L2-resident gather
    int off  = (bt * 256 + c) * 1024 + hw0 + hw;
    float zv = z[off];                                // L2-resident reload
    float d  = q - zv;
    lsum += d * d;
    zq[off] = q;                                      // forward STE value == z_q
  }
  #pragma unroll
  for (int m = 16; m >= 1; m >>= 1) lsum += __shfl_xor(lsum, m, 32);
  if (lane == 0) redLDS[w] = lsum;
  __syncthreads();
  if (t == 0)
    partials[blockIdx.x] = redLDS[0] + redLDS[1] + redLDS[2] + redLDS[3];
}

// ---------------------------------------------------------------------------
// Kernel 2: deterministic final loss reduction (1024 partials -> scalar)
// loss = beta*mean + mean = 1.25 * sum / numel
// ---------------------------------------------------------------------------
__global__ void vq_loss(const float* __restrict__ partials,
                        float* __restrict__ loss) {
  __shared__ float red[8];
  const int t = threadIdx.x;        // 256 threads
  float s = 0.f;
  #pragma unroll
  for (int i = 0; i < 4; ++i) s += partials[i * 256 + t];
  #pragma unroll
  for (int m = 16; m >= 1; m >>= 1) s += __shfl_xor(s, m, 32);
  if ((t & 31) == 0) red[t >> 5] = s;
  __syncthreads();
  if (t == 0) {
    float tot = 0.f;
    #pragma unroll
    for (int i = 0; i < 8; ++i) tot += red[i];
    *loss = 1.25f * tot / 16777216.0f;
  }
}

// ---------------------------------------------------------------------------
extern "C" void kernel_launch(void* const* d_in, const int* in_sizes, int n_in,
                              void* d_out, int out_size, void* d_ws, size_t ws_size,
                              hipStream_t stream) {
  const float* z  = (const float*)d_in[0];   // [8,8,256,32,32]
  const float* cb = (const float*)d_in[1];   // [1024,256]

  char* ws = (char*)d_ws;
  unsigned short* chi   = (unsigned short*)ws;               // 512 KB
  unsigned short* clo   = (unsigned short*)(ws + 524288);    // 512 KB
  float* enorm          = (float*)(ws + 1048576);            //   4 KB
  float* partials       = (float*)(ws + 1048576 + 4096);     //   4 KB

  float* zq      = (float*)d_out;                            // 16,777,216 f32
  float* loss    = zq + 16777216;                            // 1 f32
  int*   idx_out = (int*)d_out + 16777217;                   // 65,536 int32

  vq_prep<<<1024, 256, 0, stream>>>(cb, chi, clo, enorm);
  vq_main<<<1024, 128, 0, stream>>>(z, cb, chi, clo, enorm, zq, idx_out, partials);
  vq_loss<<<1, 256, 0, stream>>>(partials, loss);
}
